// UnsupervisedMaxSATLoss_72928544686163
// MI455X (gfx1250) — compile-verified, matrix operand order checked by reading, not run on hardware
//
#include <hip/hip_runtime.h>

// Clause-flag region: n_clauses = 4,000,000 bytes, padded to 4 MiB so the
// WMMA reduction runs over a size divisible by (4096 waves * 1024 B) with
// zero-padding contributing nothing. Counter lives right after it in d_ws.
#define FLAG_PAD (1u << 22)   // 4 MiB

typedef int v8i __attribute__((ext_vector_type(8)));
typedef int v4i __attribute__((ext_vector_type(4)));

// ---------------------------------------------------------------------------
// Kernel 1: zero the flag bytes (16B/lane, fully coalesced) + counter.
// Regular-temporal stores: these lines are immediately re-written by the
// scatter and re-read by the reduction, so we want them L2-resident.
// ---------------------------------------------------------------------------
__global__ void msat_zero(uint4* __restrict__ flags, int* __restrict__ counter) {
    const unsigned i = blockIdx.x * blockDim.x + threadIdx.x;
    const unsigned n = FLAG_PAD / 16u;
    const unsigned stride = gridDim.x * blockDim.x;
    for (unsigned j = i; j < n; j += stride) {
        flags[j] = make_uint4(0u, 0u, 0u, 0u);
    }
    if (i == 0) *counter = 0;
}

// ---------------------------------------------------------------------------
// Kernel 2: edge scatter. Streams lits/clauses as 128-bit NON-TEMPORAL loads
// (read-once data must not pollute L2 where preds/flags live), gathers preds
// (4 MB -> L2 resident), and does an idempotent byte store flags[clause] = 1
// when the literal satisfies the clause. No atomics needed: every racing
// writer stores the same constant -> deterministic.
// ---------------------------------------------------------------------------
__device__ __forceinline__ void msat_edge(int lit, int clause, int n_vars,
                                          const float* __restrict__ preds,
                                          unsigned char* __restrict__ flags) {
    const bool pos = lit < n_vars;
    const int  var = pos ? lit : (lit - n_vars);
    const bool ge  = preds[var] >= 0.5f;
    // positive literal satisfied iff pred>=0.5; negated iff pred<0.5
    if (ge == pos) flags[clause] = (unsigned char)1;
}

__global__ void msat_scatter(const float* __restrict__ preds,
                             const int*   __restrict__ lits,
                             const int*   __restrict__ clauses,
                             const int*   __restrict__ n_vars_p,
                             unsigned char* __restrict__ flags,
                             int n_edges) {
    const int n_vars = *n_vars_p;                 // uniform scalar, cached
    const int i  = blockIdx.x * blockDim.x + threadIdx.x;
    const int i4 = i * 4;
    if (i4 + 3 < n_edges) {
        const v4i l = __builtin_nontemporal_load((const v4i*)lits + i);
        const v4i c = __builtin_nontemporal_load((const v4i*)clauses + i);
        msat_edge(l.x, c.x, n_vars, preds, flags);
        msat_edge(l.y, c.y, n_vars, preds, flags);
        msat_edge(l.z, c.z, n_vars, preds, flags);
        msat_edge(l.w, c.w, n_vars, preds, flags);
    } else {
        for (int e = i4; e < n_edges; ++e) {
            msat_edge(lits[e], clauses[e], n_vars, preds, flags);
        }
    }
}

// ---------------------------------------------------------------------------
// Kernel 3: count set flag bytes with V_WMMA_I32_16X16X64_IU8.
// Per wave iteration: load 1024 contiguous flag bytes (32 B/lane via two
// non-temporal b128 loads -- last use of this data) as the 16x64 u8 A-matrix,
// multiply by an all-ones u8 B-matrix, accumulate into the i32 C/D matrix.
// Since B is all ones, D[m][n] is the row-sum of A; the matrix total
// (== popcount of the 1024 bytes) is column 0: lane 0 regs 0..7 (M=0..7)
// plus lane 16 regs 0..7 (M=8..15).
// Grid: 512 blocks x 256 threads = 4096 wave32s; 4096*1024B = FLAG_PAD, so
// every wave runs exactly one uniform iteration (EXEC all-1s at each WMMA;
// the uniform exit test only clears EXEC after the last WMMA).
// ---------------------------------------------------------------------------
__global__ void msat_reduce_wmma(const unsigned char* __restrict__ flags,
                                 int* __restrict__ counter) {
    const unsigned lane          = threadIdx.x & 31u;
    const unsigned waveInBlock   = threadIdx.x >> 5;
    const unsigned wavesPerBlock = blockDim.x >> 5;
    const unsigned waveId        = blockIdx.x * wavesPerBlock + waveInBlock;
    const unsigned nWaves        = gridDim.x * wavesPerBlock;

    v8i ones;
    v8i acc;
#pragma unroll
    for (int k = 0; k < 8; ++k) { ones[k] = 0x01010101; acc[k] = 0; }

    // Loop bound is wave-uniform: no EXEC divergence around the WMMA.
    for (unsigned base = waveId * 1024u; base < FLAG_PAD; base += nWaves * 1024u) {
        const v4i* p = (const v4i*)(flags + base) + (lane * 2u);
        const v4i a0 = __builtin_nontemporal_load(p);
        const v4i a1 = __builtin_nontemporal_load(p + 1);
        v8i A;
        A[0] = a0.x; A[1] = a0.y; A[2] = a0.z; A[3] = a0.w;
        A[4] = a1.x; A[5] = a1.y; A[6] = a1.z; A[7] = a1.w;
        // (sgn_a, A, sgn_b, B, C, reuse_a, reuse_b) — unsigned x unsigned
        acc = __builtin_amdgcn_wmma_i32_16x16x64_iu8(false, A, false, ones,
                                                     acc, false, false);
    }

    int s = acc[0] + acc[1] + acc[2] + acc[3] + acc[4] + acc[5] + acc[6] + acc[7];
    // Matrix total = column 0 = lane0 partial + lane16 partial.
    const int total = __shfl(s, 0, 32) + __shfl(s, 16, 32);
    if (lane == 0) atomicAdd(counter, total);
}

// ---------------------------------------------------------------------------
// Kernel 4: finalize scalar loss.
// ---------------------------------------------------------------------------
__global__ void msat_finalize(const int* __restrict__ counter,
                              const int* __restrict__ n_vars_p,
                              const int* __restrict__ n_clauses_p,
                              float* __restrict__ out) {
    const float n_sat = (float)(*counter);
    out[0] = ((float)(*n_clauses_p) - n_sat) / (float)(*n_vars_p);
}

// ---------------------------------------------------------------------------
extern "C" void kernel_launch(void* const* d_in, const int* in_sizes, int n_in,
                              void* d_out, int out_size, void* d_ws, size_t ws_size,
                              hipStream_t stream) {
    const float* preds       = (const float*)d_in[0];
    const int*   lits        = (const int*)  d_in[1];
    const int*   clauses     = (const int*)  d_in[2];
    const int*   n_vars_p    = (const int*)  d_in[3];
    const int*   n_clauses_p = (const int*)  d_in[4];
    const int    n_edges     = in_sizes[1];

    unsigned char* flags   = (unsigned char*)d_ws;
    int*           counter = (int*)((char*)d_ws + FLAG_PAD);

    // 1) zero flags + counter (262144 threads x 16B = 4 MiB, one pass)
    msat_zero<<<1024, 256, 0, stream>>>((uint4*)flags, counter);

    // 2) edge scatter: one thread per 4 edges
    const int t4 = (n_edges + 3) / 4;
    msat_scatter<<<(t4 + 255) / 256, 256, 0, stream>>>(
        preds, lits, clauses, n_vars_p, flags, n_edges);

    // 3) WMMA byte-sum reduction: 4096 waves x 1024 B = FLAG_PAD
    msat_reduce_wmma<<<512, 256, 0, stream>>>(flags, counter);

    // 4) scalar loss
    msat_finalize<<<1, 1, 0, stream>>>(counter, n_vars_p, n_clauses_p,
                                       (float*)d_out);
}
